// complexRNN_56221121905333
// MI455X (gfx1250) — matrix operand order repaired and checked by reading, not compiled, not run on hardware
//
#include <hip/hip_runtime.h>
#include <hip/hip_bf16.h>

// ---------------------------------------------------------------------------
// Complex RNN (T=256, B=64, H=512, L=2) for gfx1250 (MI455X).
//   Phase A (per layer): parallel input projection  acc = x (*) Wx^T + bx + bh
//     -> one big GEMM M=T*B=16384, N=512, K=512 using v_wmma_f32_16x16x32_f16
//   Phase B (per layer): serial recurrence, one kernel per timestep:
//     z = acc[t] + h (*) Wh^T ;  h = tanh(z)   (complex, split tanh)
//     M=64, N=512, K=512 per step; weights stay L2-resident.
// f16 operands, f32 WMMA accumulation.
// ---------------------------------------------------------------------------

#define TT 256
#define BB 64
#define HH 512
#define LL 2

static constexpr int    KT  = HH / 32;                  // 16 K-steps per GEMM
static constexpr size_t TBH = (size_t)TT * BB * HH;     // 8,388,608

typedef __attribute__((ext_vector_type(16))) _Float16 v16h;
typedef __attribute__((ext_vector_type(8)))  _Float16 v8h_t;
typedef __attribute__((ext_vector_type(8)))  float    v8f;

// --- A-matrix fragment load (16x32 f16, MxK). lane = m + 16*g.
// half j -> k = j + 16*g: two contiguous 16B chunks per lane.
__device__ __forceinline__ v16h load_a16(const _Float16* __restrict__ row,
                                         int k0, int g) {
  v8h_t c0 = *reinterpret_cast<const v8h_t*>(row + k0 + 16 * g);
  v8h_t c1 = *reinterpret_cast<const v8h_t*>(row + k0 + 16 * g + 8);
  v16h r;
#pragma unroll
  for (int j = 0; j < 8; ++j) { r[j] = c0[j]; r[8 + j] = c1[j]; }
  return r;
}

__device__ __forceinline__ v16h load_a32(const float* __restrict__ row,
                                         int k0, int g) {
  const float4* p0 = reinterpret_cast<const float4*>(row + k0 + 16 * g);
  const float4* p1 = reinterpret_cast<const float4*>(row + k0 + 16 * g + 8);
  float4 u0 = p0[0], u1 = p0[1], u2 = p1[0], u3 = p1[1];
  v16h r;
  r[0]  = (_Float16)u0.x; r[1]  = (_Float16)u0.y; r[2]  = (_Float16)u0.z; r[3]  = (_Float16)u0.w;
  r[4]  = (_Float16)u1.x; r[5]  = (_Float16)u1.y; r[6]  = (_Float16)u1.z; r[7]  = (_Float16)u1.w;
  r[8]  = (_Float16)u2.x; r[9]  = (_Float16)u2.y; r[10] = (_Float16)u2.z; r[11] = (_Float16)u2.w;
  r[12] = (_Float16)u3.x; r[13] = (_Float16)u3.y; r[14] = (_Float16)u3.z; r[15] = (_Float16)u3.w;
  return r;
}

// --- B-matrix fragment load (32x16 f16, KxN), B[k][n] = W[n][k].
// lane = n + 16*(k/16), slots j = k%16: 16 contiguous f16 (32B) from row n of W.
__device__ __forceinline__ v16h load16h(const _Float16* __restrict__ p) {
  v8h_t lo = *reinterpret_cast<const v8h_t*>(p);
  v8h_t hi = *reinterpret_cast<const v8h_t*>(p + 8);
  v16h r;
#pragma unroll
  for (int j = 0; j < 8; ++j) { r[j] = lo[j]; r[8 + j] = hi[j]; }
  return r;
}

#define WMMA_F16(A, B, C) \
  __builtin_amdgcn_wmma_f32_16x16x32_f16(false, (A), false, (B), (short)0, (C), false, false)

// ---------------------------------------------------------------------------
// Prep: convert the 8 weight matrices to f16 (row-major passthrough) and
// fold the two biases per component into one.
// ---------------------------------------------------------------------------
__global__ __launch_bounds__(256)
void prep_kernel(const float* __restrict__ Wxr, const float* __restrict__ Wxi,
                 const float* __restrict__ Whr, const float* __restrict__ Whi,
                 const float* __restrict__ bxr, const float* __restrict__ bxi,
                 const float* __restrict__ bhr, const float* __restrict__ bhi,
                 _Float16* __restrict__ wf16,
                 float* __restrict__ biasR, float* __restrict__ biasI) {
  const int idx = blockIdx.x * blockDim.x + threadIdx.x;
  const int WELEMS = LL * 4 * HH * HH;
  if (idx < WELEMS) {
    const int mat = idx / (HH * HH);      // l*4 + m
    const int rem = idx % (HH * HH);
    const int l = mat >> 2, m = mat & 3;
    const float* src = (m == 0) ? Wxr : (m == 1) ? Wxi : (m == 2) ? Whr : Whi;
    wf16[idx] = (_Float16)src[(size_t)l * HH * HH + rem];
  }
  if (idx < 2 * LL * HH) {
    const int comp = idx / (LL * HH);
    const int r    = idx % (LL * HH);
    if (comp == 0) biasR[r] = bxr[r] + bhr[r];
    else           biasI[r] = bxi[r] + bhi[r];
  }
}

__global__ __launch_bounds__(256)
void inith_kernel(const float* __restrict__ initR, const float* __restrict__ initI,
                  _Float16* __restrict__ hR, _Float16* __restrict__ hI) {
  const int i = blockIdx.x * blockDim.x + threadIdx.x;
  if (i < BB * HH) {
    hR[i] = (_Float16)initR[i];
    hI[i] = (_Float16)initI[i];
  }
}

// ---------------------------------------------------------------------------
// Phase A: acc_r = xr@Wxr^T - xi@Wxi^T + (bxr+bhr)
//          acc_i = xi@Wxr^T + xr@Wxi^T + (bxi+bhi)
// Grid: 1024 blocks (one 16-row M-tile each) x 8 waves (one 64-col N-strip).
// Each wave: 4 N-tiles x {accR, accI}, K-loop of 16 x 4 WMMAs.
// ---------------------------------------------------------------------------
template <bool IN_F16>
__global__ __launch_bounds__(256)
void proj_kernel(const float* __restrict__ xr32, const float* __restrict__ xi32,
                 const _Float16* __restrict__ xr16, const _Float16* __restrict__ xi16,
                 const _Float16* __restrict__ wxr, const _Float16* __restrict__ wxi,
                 const float* __restrict__ biasR, const float* __restrict__ biasI,
                 float* __restrict__ accR, float* __restrict__ accI) {
  const int lane   = threadIdx.x & 31;
  const int wave   = threadIdx.x >> 5;   // N-strip 0..7
  const int m_tile = blockIdx.x;         // 0..1023  (rows = flattened t*B+b)
  const int g      = lane >> 4;
  const int l15    = lane & 15;
  const int rA     = m_tile * 16 + l15;  // A-matrix row for this lane
  const int nb0    = wave * 64;

  v8f zero = {};
  v8f aR[4] = {zero, zero, zero, zero};
  v8f aI[4] = {zero, zero, zero, zero};

  for (int ks = 0; ks < KT; ++ks) {
    const int k0 = ks * 32;
    v16h ar, ai;
    if constexpr (IN_F16) {
      ar = load_a16(xr16 + (size_t)rA * HH, k0, g);
      ai = load_a16(xi16 + (size_t)rA * HH, k0, g);
    } else {
      ar = load_a32(xr32 + (size_t)rA * HH, k0, g);
      ai = load_a32(xi32 + (size_t)rA * HH, k0, g);
    }
    const v16h ain = -ai;                // f16 WMMA has no A/B NEG bits
    const int  bk  = k0 + (g << 4);
#pragma unroll
    for (int j = 0; j < 4; ++j) {
      const int nb = nb0 + j * 16;
      v16h br = load16h(wxr + (size_t)(nb + l15) * HH + bk);
      v16h bi = load16h(wxi + (size_t)(nb + l15) * HH + bk);
      aR[j] = WMMA_F16(ar,  br, aR[j]);
      aR[j] = WMMA_F16(ain, bi, aR[j]);
      aI[j] = WMMA_F16(ai,  br, aI[j]);
      aI[j] = WMMA_F16(ar,  bi, aI[j]);
    }
  }

  // C/D layout: slot v -> m_local = v + 8*g ; n_local = lane&15
#pragma unroll
  for (int j = 0; j < 4; ++j) {
    const int   col = nb0 + j * 16 + l15;
    const float br_ = biasR[col];
    const float bi_ = biasI[col];
#pragma unroll
    for (int v = 0; v < 8; ++v) {
      const int    m   = m_tile * 16 + v + 8 * g;
      const size_t idx = (size_t)m * HH + col;
      accR[idx] = aR[j][v] + br_;
      accI[idx] = aI[j][v] + bi_;
    }
  }
}

// ---------------------------------------------------------------------------
// Phase B, one timestep: z = acc[t] + h (*) Wh^T ; h = tanh(z).
// Grid: 16 blocks x 8 waves = 128 waves; wave -> (m_tile 0..3, n_tile 0..31).
// Ping-pong f16 hidden-state buffers avoid cross-workgroup RAW hazards.
// NOTE: outR/outI may alias accR/accI (in-place tanh over the accumulator
// plane for the last layer) -> no __restrict__ on those four.
// ---------------------------------------------------------------------------
__global__ __launch_bounds__(256)
void recur_kernel(const _Float16* __restrict__ hinR, const _Float16* __restrict__ hinI,
                  _Float16* __restrict__ houtR, _Float16* __restrict__ houtI,
                  const _Float16* __restrict__ whr, const _Float16* __restrict__ whi,
                  const float* accR, const float* accI,
                  float* outR, float* outI,
                  _Float16* __restrict__ sR16, _Float16* __restrict__ sI16,
                  float* __restrict__ finR, float* __restrict__ finI,
                  int t) {
  const int lane   = threadIdx.x & 31;
  const int wave   = threadIdx.x >> 5;
  const int W      = blockIdx.x * 8 + wave;   // 0..127
  const int m_tile = W >> 5;                  // 0..3
  const int n_tile = W & 31;                  // 0..31
  const int g      = lane >> 4;
  const int l15    = lane & 15;
  const int b_row  = m_tile * 16 + l15;

  v8f zero = {};
  v8f aR = zero, aI = zero;

  const _Float16* hrRow = hinR + (size_t)b_row * HH;
  const _Float16* hiRow = hinI + (size_t)b_row * HH;
  const size_t wOff = (size_t)(n_tile * 16 + l15) * HH + (g << 4);
  const _Float16* wrP = whr + wOff;
  const _Float16* wiP = whi + wOff;

  for (int ks = 0; ks < KT; ++ks) {
    const int k0 = ks * 32;
    v16h ar  = load_a16(hrRow, k0, g);
    v16h ai  = load_a16(hiRow, k0, g);
    v16h ain = -ai;
    v16h br  = load16h(wrP + k0);
    v16h bi  = load16h(wiP + k0);
    if (ks + 1 < KT) {                       // global_prefetch_b8 next K-slab
      __builtin_prefetch(wrP + k0 + 32, 0, 1);
      __builtin_prefetch(wiP + k0 + 32, 0, 1);
    }
    aR = WMMA_F16(ar,  br, aR);
    aR = WMMA_F16(ain, bi, aR);
    aI = WMMA_F16(ai,  br, aI);
    aI = WMMA_F16(ar,  bi, aI);
  }

  const int    col  = n_tile * 16 + l15;
  const size_t accT = (size_t)t * BB * HH;
#pragma unroll
  for (int v = 0; v < 8; ++v) {
    const int    b    = m_tile * 16 + v + 8 * g;
    const size_t hIdx = (size_t)b * HH + col;
    const size_t idx  = accT + hIdx;
    const float zr = aR[v] + accR[idx];
    const float zi = aI[v] + accI[idx];
    const float hr = tanhf(zr);
    const float hi = tanhf(zi);
    houtR[hIdx] = (_Float16)hr;              // next step's A-matrix
    houtI[hIdx] = (_Float16)hi;
    if (sR16) { sR16[idx] = (_Float16)hr; sI16[idx] = (_Float16)hi; }  // feed layer l+1
    if (outR) { outR[idx] = hr; outI[idx] = hi; }                      // final f32 seq
    if (finR) { finR[hIdx] = hr; finI[hIdx] = hi; }                    // final hidden
  }
}

// ---------------------------------------------------------------------------
extern "C" void kernel_launch(void* const* d_in, const int* in_sizes, int n_in,
                              void* d_out, int out_size, void* d_ws, size_t ws_size,
                              hipStream_t stream) {
  (void)in_sizes; (void)n_in; (void)out_size; (void)ws_size;

  const float* seq_r  = (const float*)d_in[0];
  const float* seq_i  = (const float*)d_in[1];
  const float* init_r = (const float*)d_in[2];
  const float* init_i = (const float*)d_in[3];
  const float* Wxr    = (const float*)d_in[4];
  const float* Wxi    = (const float*)d_in[5];
  const float* Whr    = (const float*)d_in[6];
  const float* Whi    = (const float*)d_in[7];
  const float* bxr    = (const float*)d_in[8];
  const float* bxi    = (const float*)d_in[9];
  const float* bhr    = (const float*)d_in[10];
  const float* bhi    = (const float*)d_in[11];

  float* out  = (float*)d_out;
  float* accR = out;                               // [T*B,H]; becomes seq_r f32
  float* accI = out + TBH;                         // becomes seq_i f32
  float* finR = out + 2 * TBH;                     // [L,B,H]
  float* finI = finR + (size_t)LL * BB * HH;

  // ---- workspace carve-up (~36.5 MB) ----
  char*  ws  = (char*)d_ws;
  size_t off = 0;
  _Float16* wf16 = (_Float16*)(ws + off); off += (size_t)LL * 4 * HH * HH * sizeof(_Float16);
  float* biasR   = (float*)(ws + off);    off += (size_t)LL * HH * sizeof(float);
  float* biasI   = (float*)(ws + off);    off += (size_t)LL * HH * sizeof(float);
  off = (off + 255) & ~(size_t)255;
  _Float16* hbase = (_Float16*)(ws + off); off += (size_t)4 * BB * HH * sizeof(_Float16);
  off = (off + 255) & ~(size_t)255;
  _Float16* seqR16 = (_Float16*)(ws + off); off += TBH * sizeof(_Float16);
  _Float16* seqI16 = (_Float16*)(ws + off); off += TBH * sizeof(_Float16);

  _Float16* hR[2] = { hbase,                      hbase + 2 * (size_t)BB * HH };
  _Float16* hI[2] = { hbase + (size_t)BB * HH,    hbase + 3 * (size_t)BB * HH };

  const size_t WMAT = (size_t)HH * HH;

  prep_kernel<<<(LL * 4 * HH * HH) / 256, 256, 0, stream>>>(
      Wxr, Wxi, Whr, Whi, bxr, bxi, bhr, bhi, wf16, biasR, biasI);

  for (int l = 0; l < LL; ++l) {
    const _Float16* wxr16 = wf16 + (size_t)(4 * l + 0) * WMAT;
    const _Float16* wxi16 = wf16 + (size_t)(4 * l + 1) * WMAT;
    const _Float16* whr16 = wf16 + (size_t)(4 * l + 2) * WMAT;
    const _Float16* whi16 = wf16 + (size_t)(4 * l + 3) * WMAT;

    inith_kernel<<<(BB * HH + 255) / 256, 256, 0, stream>>>(
        init_r + (size_t)l * BB * HH, init_i + (size_t)l * BB * HH, hR[0], hI[0]);

    if (l == 0) {
      proj_kernel<false><<<(TT * BB) / 16, 256, 0, stream>>>(
          seq_r, seq_i, nullptr, nullptr, wxr16, wxi16,
          biasR + l * HH, biasI + l * HH, accR, accI);
    } else {
      proj_kernel<true><<<(TT * BB) / 16, 256, 0, stream>>>(
          nullptr, nullptr, seqR16, seqI16, wxr16, wxi16,
          biasR + l * HH, biasI + l * HH, accR, accI);
    }

    const bool last = (l == LL - 1);
    for (int t = 0; t < TT; ++t) {
      recur_kernel<<<16, 256, 0, stream>>>(
          hR[t & 1], hI[t & 1], hR[(t + 1) & 1], hI[(t + 1) & 1],
          whr16, whi16, accR, accI,
          last ? accR : nullptr, last ? accI : nullptr,
          last ? nullptr : seqR16, last ? nullptr : seqI16,
          (t == TT - 1) ? finR + (size_t)l * BB * HH : nullptr,
          (t == TT - 1) ? finI + (size_t)l * BB * HH : nullptr,
          t);
    }
  }
}
